// ConditionalFeedForward_489626272252
// MI455X (gfx1250) — matrix operand order repaired and checked by reading, not compiled, not run on hardware
//
#include <hip/hip_runtime.h>
#include <math.h>

// Problem constants (from reference): T=512, TOPK=2, E=8, H=2816, D=1024
#define TT    512
#define TOPK  2
#define NP    (TT * TOPK)     // 1024 (token, expert) assignment slots
#define NE    8
#define HD    2816
#define DD    1024
#define PB    16              // 64-row blocks per expert (worst case 1024 rows -> 16 blocks)

#define GLOBAL __attribute__((address_space(1)))

typedef __attribute__((ext_vector_type(16))) _Float16 v16h;
typedef __attribute__((ext_vector_type(8)))  _Float16 v8h;
typedef __attribute__((ext_vector_type(8)))  float    v8f;
typedef __attribute__((ext_vector_type(4)))  float    v4f;

// ---- operand loaders (explicit global address space => global_load_*) ---
// B operand (32x16, K x N): lane n (0..15) holds column n with K = 0..15,
// lane n+16 holds K = 16..31 (contiguous 16 K-values per lane).
__device__ __forceinline__ v16h load_b_krow(const GLOBAL float* p) {
  v4f f0 = *(const GLOBAL v4f*)(p + 0);
  v4f f1 = *(const GLOBAL v4f*)(p + 4);
  v4f f2 = *(const GLOBAL v4f*)(p + 8);
  v4f f3 = *(const GLOBAL v4f*)(p + 12);
  v16h r;
#pragma unroll
  for (int j = 0; j < 4; ++j) {
    r[j]      = (_Float16)f0[j];
    r[4 + j]  = (_Float16)f1[j];
    r[8 + j]  = (_Float16)f2[j];
    r[12 + j] = (_Float16)f3[j];
  }
  return r;
}

// A operand (16x32, M x K): lane m holds K = kb..kb+15,
// kb = 16*(lane>=16). p points at (row, k0+kb); chunks at +0 and +8.
__device__ __forceinline__ v16h load_a_row(const GLOBAL _Float16* p) {
  v8h lo = *(const GLOBAL v8h*)(p);
  v8h hi = *(const GLOBAL v8h*)(p + 8);
  v16h r;
#pragma unroll
  for (int j = 0; j < 8; ++j) { r[j] = lo[j]; r[8 + j] = hi[j]; }
  return r;
}

__device__ __forceinline__ v8f wmma16(v16h a, v16h b, v8f c) {
  return __builtin_amdgcn_wmma_f32_16x16x32_f16(false, a, false, b, (short)0,
                                                c, false, false);
}

// ---- routing: stable counting sort of the 1024 assignment slots --------
__global__ void moe_route(const int* __restrict__ eidx,
                          int* __restrict__ off, int* __restrict__ perm) {
  __shared__ int cnt[NE];
  __shared__ int soff[NE + 1];
  int tid = threadIdx.x;
  if (tid < NE) cnt[tid] = 0;
  __syncthreads();
  for (int s = tid; s < NP; s += blockDim.x) atomicAdd(&cnt[eidx[s]], 1);
  __syncthreads();
  if (tid == 0) {
    int a = 0;
    for (int e = 0; e < NE; ++e) { soff[e] = a; a += cnt[e]; }
    soff[NE] = a;
    for (int e = 0; e <= NE; ++e) off[e] = soff[e];
  }
  __syncthreads();
  if (tid < NE) {  // stable placement, one thread per expert
    int w = soff[tid];
    for (int s = 0; s < NP; ++s)
      if (eidx[s] == tid) perm[w++] = s;
  }
}

// ---- x: f32 -> f16 -----------------------------------------------------
__global__ void cvt_x(const float* __restrict__ x, _Float16* __restrict__ xb) {
  int i = blockIdx.x * blockDim.x + threadIdx.x;
  if (i < TT * DD) xb[i] = (_Float16)x[i];
}

// ---- gate kernel body, specialized on row-tile count -------------------
template <int NRT>
__device__ __forceinline__ void gate_body(
    const GLOBAL _Float16* __restrict__ xb, const GLOBAL float* __restrict__ p1,
    const GLOBAL float* __restrict__ p3, const int* slt, const int* tok,
    int lane, int col, int nrows, int hrow, GLOBAL _Float16* __restrict__ hws) {
  int kbA = (lane >> 4) * 16;
  const GLOBAL _Float16* ap[NRT];
#pragma unroll
  for (int rt = 0; rt < NRT; ++rt)
    ap[rt] = xb + (size_t)tok[rt * 16 + col] * DD + kbA;

  v8f acc1[NRT] = {}; v8f acc3[NRT] = {};
  for (int k0 = 0; k0 < DD; k0 += 32) {
    v16h b1 = load_b_krow(p1); p1 += 32;
    v16h b3 = load_b_krow(p3); p3 += 32;
#pragma unroll
    for (int rt = 0; rt < NRT; ++rt) {
      v16h a = load_a_row(ap[rt]); ap[rt] += 32;
      acc1[rt] = wmma16(a, b1, acc1[rt]);
      acc3[rt] = wmma16(a, b3, acc3[rt]);
    }
  }

  int mhi = (lane >> 4) * 8;                 // C layout: M = j + 8*(lane>=16)
#pragma unroll
  for (int rt = 0; rt < NRT; ++rt) {
#pragma unroll
    for (int j = 0; j < 8; ++j) {
      int m = rt * 16 + mhi + j;
      if (m < nrows) {
        float z1 = acc1[rt][j], z3 = acc3[rt][j];
        // silu(z1)*z3 with fast reciprocal (v_rcp_f32); stored as f16 anyway
        float hv = z1 * __builtin_amdgcn_rcpf(1.f + __expf(-z1)) * z3;
        hws[(size_t)slt[m] * HD + hrow] = (_Float16)hv;
      }
    }
  }
}

// grid: (NE*PB, HD/64); block 128 (4 waves x 16 H-columns).
__global__ __launch_bounds__(128) void moe_gate(
    const _Float16* __restrict__ xb_, const float* __restrict__ w1,
    const float* __restrict__ w3, const int* __restrict__ off,
    const int* __restrict__ perm, _Float16* __restrict__ hws_) {
  int e = blockIdx.x / PB, g = blockIdx.x % PB;
  int r0 = off[e] + g * 64, r1 = off[e + 1];
  if (r0 >= r1) return;                      // block-uniform early exit
  int nrows = min(64, r1 - r0);
  __shared__ int slt[64], tok[64];
  int tid = threadIdx.x;
  if (tid < 64) {
    int s = perm[min(r0 + tid, r1 - 1)];     // clamp pad rows (stores guarded)
    slt[tid] = s;
    tok[tid] = s / TOPK;
  }
  __syncthreads();

  int lane = tid & 31;
  int col  = lane & 15;
  int hrow = blockIdx.y * 64 + (tid >> 5) * 16 + col;  // w1/w3 row == H column
  int kb16 = (lane >> 4) * 16;
  const GLOBAL float* p1 =
      (const GLOBAL float*)(w1 + ((size_t)e * HD + hrow) * DD + kb16);
  const GLOBAL float* p3 =
      (const GLOBAL float*)(w3 + ((size_t)e * HD + hrow) * DD + kb16);
  const GLOBAL _Float16* xb = (const GLOBAL _Float16*)xb_;
  GLOBAL _Float16* hws = (GLOBAL _Float16*)hws_;

  switch ((nrows + 15) >> 4) {               // block-uniform dispatch
    case 1: gate_body<1>(xb, p1, p3, slt, tok, lane, col, nrows, hrow, hws); break;
    case 2: gate_body<2>(xb, p1, p3, slt, tok, lane, col, nrows, hrow, hws); break;
    case 3: gate_body<3>(xb, p1, p3, slt, tok, lane, col, nrows, hrow, hws); break;
    default: gate_body<4>(xb, p1, p3, slt, tok, lane, col, nrows, hrow, hws); break;
  }
}

// ---- down kernel body, specialized on row-tile count -------------------
template <int NRT>
__device__ __forceinline__ void down_body(
    const GLOBAL _Float16* __restrict__ hws, const GLOBAL float* __restrict__ wq,
    const int* slt, int lane, int col, int nrows, int icol,
    GLOBAL float* __restrict__ out) {
  int kbA = (lane >> 4) * 16;
  const GLOBAL _Float16* ap[NRT];
#pragma unroll
  for (int rt = 0; rt < NRT; ++rt)
    ap[rt] = hws + (size_t)slt[rt * 16 + col] * HD + kbA;

  v8f acc[NRT] = {};
  for (int k0 = 0; k0 < HD; k0 += 32) {
    v16h b;
#pragma unroll
    for (int j = 0; j < 16; ++j)             // constant j*DD immediate offsets
      b[j] = (_Float16)wq[(size_t)j * DD];
    wq += (size_t)32 * DD;
#pragma unroll
    for (int rt = 0; rt < NRT; ++rt) {
      v16h a = load_a_row(ap[rt]); ap[rt] += 32;
      acc[rt] = wmma16(a, b, acc[rt]);
    }
  }

  int mhi = (lane >> 4) * 8;
#pragma unroll
  for (int rt = 0; rt < NRT; ++rt) {
#pragma unroll
    for (int j = 0; j < 8; ++j) {
      int m = rt * 16 + mhi + j;
      if (m < nrows) out[(size_t)slt[m] * DD + icol] = acc[rt][j];
    }
  }
}

// grid: (NE*PB, DD/64); block 128. w2 column loads: 16 lanes x 4B = 64B/row.
__global__ __launch_bounds__(128) void moe_down(
    const _Float16* __restrict__ hws_, const float* __restrict__ w2,
    const int* __restrict__ off, const int* __restrict__ perm,
    float* __restrict__ out_) {
  int e = blockIdx.x / PB, g = blockIdx.x % PB;
  int r0 = off[e] + g * 64, r1 = off[e + 1];
  if (r0 >= r1) return;
  int nrows = min(64, r1 - r0);
  __shared__ int slt[64];
  int tid = threadIdx.x;
  if (tid < 64) slt[tid] = perm[min(r0 + tid, r1 - 1)];
  __syncthreads();

  int lane = tid & 31;
  int col  = lane & 15;
  int icol = blockIdx.y * 64 + (tid >> 5) * 16 + col;  // output D column
  int kb16 = (lane >> 4) * 16;
  // advancing base: element (k0+kb16+j, icol) = wq[(k0+j)*DD] after hoisting kb16
  const GLOBAL float* wq =
      (const GLOBAL float*)(w2 + (size_t)e * HD * DD + (size_t)kb16 * DD + icol);
  const GLOBAL _Float16* hws = (const GLOBAL _Float16*)hws_;
  GLOBAL float* out = (GLOBAL float*)out_;

  switch ((nrows + 15) >> 4) {
    case 1: down_body<1>(hws, wq, slt, lane, col, nrows, icol, out); break;
    case 2: down_body<2>(hws, wq, slt, lane, col, nrows, icol, out); break;
    case 3: down_body<3>(hws, wq, slt, lane, col, nrows, icol, out); break;
    default: down_body<4>(hws, wq, slt, lane, col, nrows, icol, out); break;
  }
}

extern "C" void kernel_launch(void* const* d_in, const int* in_sizes, int n_in,
                              void* d_out, int out_size, void* d_ws, size_t ws_size,
                              hipStream_t stream) {
  (void)in_sizes; (void)n_in; (void)out_size; (void)ws_size;
  const float* x    = (const float*)d_in[0];
  const int*   eidx = (const int*)d_in[1];
  const float* w1   = (const float*)d_in[2];
  const float* w2   = (const float*)d_in[3];
  const float* w3   = (const float*)d_in[4];
  float* out = (float*)d_out;

  // workspace layout: [off 16 ints][perm 1024 ints][pad to 8KB][xb f16 1MB][hws f16 5.5MB]
  char* ws = (char*)d_ws;
  int* off  = (int*)ws;
  int* perm = (int*)(ws + 64);
  _Float16* xb  = (_Float16*)(ws + 8192);
  _Float16* hws = (_Float16*)(ws + 8192 + (size_t)TT * DD * sizeof(_Float16));

  moe_route<<<1, 256, 0, stream>>>(eidx, off, perm);
  cvt_x<<<(TT * DD) / 256, 256, 0, stream>>>(x, xb);
  moe_gate<<<dim3(NE * PB, HD / 64), 128, 0, stream>>>(xb, w1, w3, off, perm, hws);
  moe_down<<<dim3(NE * PB, DD / 64), 128, 0, stream>>>(hws, w2, off, perm, out);
}
